// RobertaGNNModel_42820823941326
// MI455X (gfx1250) — matrix operand order repaired and checked by reading, not compiled, not run on hardware
//
#include <hip/hip_runtime.h>
#include <math.h>

// ---------------- problem constants ----------------
#define Bc   32
#define Sc   128
#define Hc   768
#define Lc   12
#define NHc  12
#define DHc  64
#define FFc  3072
#define Nc   (Bc * Sc)          // 4096
#define EDc  16
#define Ec   65536
#define E2c  (Ec + Nc)          // 69632

typedef __bf16 bf16;
typedef __attribute__((ext_vector_type(16))) __bf16 v16bf;
typedef __attribute__((ext_vector_type(8)))  float  v8f;

union FragBF { v16bf v; unsigned u[8]; };

static __device__ inline v8f wmma_bf16(v16bf a, v16bf b, v8f c) {
  return __builtin_amdgcn_wmma_f32_16x16x32_bf16(
      /*neg_a=*/false, a, /*neg_b=*/false, b,
      /*c_mod=*/(short)0, c, /*reuse_a=*/false, /*reuse_b=*/false);
}

// low 32 bits of a generic pointer to LDS == wave-relative LDS byte address
// (ISA 10.2)
static __device__ inline unsigned lds_addr32(const void* p) {
  return (unsigned)(unsigned long long)p;
}

// async DMA: 16 bytes global -> LDS per active lane (ASYNCcnt-tracked)
static __device__ inline void async_copy_b128(unsigned lds_off, const void* gptr) {
  asm volatile("global_load_async_to_lds_b128 %0, %1, off"
               :: "v"(lds_off), "v"(gptr)
               : "memory");
}
static __device__ inline void wait_async0() {
  asm volatile("s_wait_asynccnt 0" ::: "memory");
}

// A-matrix 16x32 bf16 fragment from row-major LDS [rows][ld].
// Lane layout (ISA 7.12.2): m = lane&15, half = lane>>4.
// VGPR v in 0..7: K = half*16 + 2v,2v+1 (contiguous, matches load_frag_b).
static __device__ inline v16bf load_frag_a(const bf16* base, int row0, int ld,
                                           int k0, int lane) {
  int m = lane & 15, hl = lane >> 4;
  const bf16* r = base + (size_t)(row0 + m) * ld + k0;
  FragBF f;
#pragma unroll
  for (int v = 0; v < 8; ++v) f.u[v] = *(const unsigned*)(r + hl * 16 + 2 * v);
  return f.v;
}

// B-matrix 32x16 bf16 fragment. B kept *transposed* in LDS: baseT[n][k],
// row stride ldk; lane (n = lane&15, half = lane>>4) reads contiguous K:
// element e -> K = half*16 + e.
static __device__ inline v16bf load_frag_b(const bf16* baseT, int n0, int ldk,
                                           int k0, int lane) {
  int n = lane & 15, hl = lane >> 4;
  const bf16* r = baseT + (size_t)(n0 + n) * ldk + k0 + hl * 16;
  FragBF f;
#pragma unroll
  for (int v = 0; v < 8; ++v) f.u[v] = *(const unsigned*)(r + 2 * v);
  return f.v;
}

static __device__ inline float gelu_f(float x) {
  return 0.5f * x * (1.0f + erff(x * 0.70710678118654752f));
}

// monotone float<->uint key for atomic max over signed floats
static __device__ inline unsigned fenc(float x) {
  unsigned u = __float_as_uint(x);
  return (u & 0x80000000u) ? ~u : (u | 0x80000000u);
}
static __device__ inline float fdec(unsigned k) {
  return (k & 0x80000000u) ? __uint_as_float(k & 0x7fffffffu)
                           : __uint_as_float(~k);
}

// ---------------- small utility kernels ----------------
__global__ void k_convert_bf16(const float* __restrict__ src, bf16* __restrict__ dst, size_t n) {
  size_t i = (size_t)blockIdx.x * blockDim.x + threadIdx.x;
  if (i < n) dst[i] = (bf16)src[i];
}
__global__ void k_zero_f32(float* p, size_t n) {
  size_t i = (size_t)blockIdx.x * blockDim.x + threadIdx.x;
  if (i < n) p[i] = 0.0f;
}
__global__ void k_zero_u32(unsigned* p, size_t n) {
  size_t i = (size_t)blockIdx.x * blockDim.x + threadIdx.x;
  if (i < n) p[i] = 0u;
}

__global__ void k_embed(const int* __restrict__ ids, const float* __restrict__ word,
                        const float* __restrict__ pos, const float* __restrict__ tok,
                        float* __restrict__ out) {
  size_t i = (size_t)blockIdx.x * blockDim.x + threadIdx.x;
  if (i >= (size_t)Nc * Hc) return;
  int n = (int)(i / Hc), h = (int)(i % Hc);
  int id = ids[n];
  int p = (n % Sc) + 2;
  out[i] = word[(size_t)id * Hc + h] + pos[(size_t)p * Hc + h] + tok[h];
}

// LayerNorm over last dim; writes f32 and (optionally) bf16
__global__ __launch_bounds__(256) void k_layernorm(const float* __restrict__ x,
                                                   const float* __restrict__ g,
                                                   const float* __restrict__ b,
                                                   float* __restrict__ y,
                                                   bf16* __restrict__ ybf, int Hdim) {
  int row = blockIdx.x;
  const float* xr = x + (size_t)row * Hdim;
  float s = 0.f, s2 = 0.f;
  for (int h = threadIdx.x; h < Hdim; h += 256) {
    float v = xr[h];
    s += v; s2 += v * v;
  }
  for (int off = 1; off < 32; off <<= 1) {
    s  += __shfl_xor(s,  off, 32);
    s2 += __shfl_xor(s2, off, 32);
  }
  __shared__ float rs[8], rs2[8];
  int w = threadIdx.x >> 5, lane = threadIdx.x & 31;
  if (!lane) { rs[w] = s; rs2[w] = s2; }
  __syncthreads();
  if (threadIdx.x == 0) {
    float a = 0.f, a2 = 0.f;
#pragma unroll
    for (int i = 0; i < 8; ++i) { a += rs[i]; a2 += rs2[i]; }
    rs[0] = a; rs2[0] = a2;
  }
  __syncthreads();
  float mean = rs[0] / Hdim;
  float var = rs2[0] / Hdim - mean * mean;
  float inv = rsqrtf(var + 1e-5f);
  for (int h = threadIdx.x; h < Hdim; h += 256) {
    float v = (xr[h] - mean) * inv * g[h] + b[h];
    y[(size_t)row * Hdim + h] = v;
    if (ybf) ybf[(size_t)row * Hdim + h] = (bf16)v;
  }
}

// ---------------- WMMA GEMM ----------------
// C[M,N] = act( A[M,K](bf16) x B[K,N](bf16) + bias[N] [+ resid] )
// 256 threads = 8 waves (4 M-waves x 2 N-waves). Block tile 128x128, K step 32.
// Wave tile 32x64 -> 2x4 fragments -> 8 WMMA per K step.
// A tile DMA'd via global_load_async_to_lds_b128 (ASYNCcnt); B tile transposed
// through VGPRs into LDS [n][k] so WMMA fragments read contiguous K pairs.
#define TM 128
#define TN 128
#define TK 32
template <int ACT, bool RES, bool OUTF, bool OUTB>
__global__ __launch_bounds__(256) void gemm_bf16_kernel(
    const bf16* __restrict__ A, const bf16* __restrict__ Bw,
    const float* __restrict__ bias, const float* __restrict__ resid,
    float* __restrict__ Cf, bf16* __restrict__ Cb,
    int M, int Nn, int K) {
  __shared__ __align__(16) bf16 As[TM * TK];   // [128][32] row-major
  __shared__ __align__(16) bf16 BsT[TN * TK];  // [128][32] transposed: [n][k]
  int tid = threadIdx.x;
  int lane = tid & 31, w = tid >> 5;
  int wm = w >> 1, wn = w & 1;
  int mblk = blockIdx.y * TM, nblk = blockIdx.x * TN;
  unsigned asBase = lds_addr32(As);

  v8f acc[2][4];
#pragma unroll
  for (int im = 0; im < 2; ++im)
#pragma unroll
    for (int in = 0; in < 4; ++in)
      acc[im][in] = (v8f){0.f, 0.f, 0.f, 0.f, 0.f, 0.f, 0.f, 0.f};

  for (int k0 = 0; k0 < K; k0 += TK) {
    // A tile: 128x32 bf16 = 8 KB = 512 x 16B chunks, async DMA to LDS
#pragma unroll
    for (int i = tid; i < 512; i += 256) {
      int r = i >> 2, c = i & 3;                 // 4 chunks per row
      int gr = mblk + r; if (gr >= M) gr = M - 1;
      async_copy_b128(asBase + (unsigned)(r * 64 + c * 16),
                      A + (size_t)gr * K + k0 + c * 8);
    }
    // B tile 32x128 (row-major global), stored transposed in LDS
#pragma unroll
    for (int i = tid; i < (TK * TN) / 2; i += 256) {  // 2048 dwords
      int kk = i >> 6, n2 = i & 63;
      unsigned uv = *(const unsigned*)(Bw + (size_t)(k0 + kk) * Nn + nblk + n2 * 2);
      bf16 lo = ((const bf16*)&uv)[0];
      bf16 hi = ((const bf16*)&uv)[1];
      BsT[(size_t)(n2 * 2) * TK + kk] = lo;
      BsT[(size_t)(n2 * 2 + 1) * TK + kk] = hi;
    }
    wait_async0();
    __syncthreads();
    v16bf af[2], bfr[4];
#pragma unroll
    for (int im = 0; im < 2; ++im)
      af[im] = load_frag_a(As, wm * 32 + im * 16, TK, 0, lane);
#pragma unroll
    for (int in = 0; in < 4; ++in)
      bfr[in] = load_frag_b(BsT, wn * 64 + in * 16, TK, 0, lane);
#pragma unroll
    for (int im = 0; im < 2; ++im)
#pragma unroll
      for (int in = 0; in < 4; ++in)
        acc[im][in] = wmma_bf16(af[im], bfr[in], acc[im][in]);
    __syncthreads();
  }

  int hl = lane >> 4, nl = lane & 15;
#pragma unroll
  for (int im = 0; im < 2; ++im) {
#pragma unroll
    for (int in = 0; in < 4; ++in) {
      int gn = nblk + wn * 64 + in * 16 + nl;
#pragma unroll
      for (int r = 0; r < 8; ++r) {
        int gm = mblk + wm * 32 + im * 16 + hl * 8 + r;
        if (gm < M) {
          float v = acc[im][in][r] + bias[gn];
          if (RES) v += resid[(size_t)gm * Nn + gn];
          if (ACT == 1) v = gelu_f(v);
          else if (ACT == 2) v = fmaxf(v, 0.f);
          if (OUTF) Cf[(size_t)gm * Nn + gn] = v;
          if (OUTB) Cb[(size_t)gm * Nn + gn] = (bf16)v;
        }
      }
    }
  }
}

template <int ACT, bool RES, bool OUTF, bool OUTB>
static void launch_gemm(const bf16* A, const bf16* Bw, const float* bias,
                        const float* resid, float* Cf, bf16* Cb,
                        int M, int Nn, int K, hipStream_t stream) {
  dim3 g(Nn / TN, (M + TM - 1) / TM);
  gemm_bf16_kernel<ACT, RES, OUTF, OUTB><<<g, 256, 0, stream>>>(
      A, Bw, bias, resid, Cf, Cb, M, Nn, K);
}

// ---------------- fused attention (one block per (batch, head)) ----------------
// dyn LDS: Qs[128*64] + Ks[128*64] + VsT[64*128] + Ps[128*128] bf16 = 80 KB
__global__ __launch_bounds__(256) void attn_kernel(
    const bf16* __restrict__ Qb, const bf16* __restrict__ Kb,
    const bf16* __restrict__ Vb, const float* __restrict__ maskp,
    bf16* __restrict__ ctx) {
  extern __shared__ char smem[];
  bf16* Qs  = (bf16*)smem;                  // [128][64]
  bf16* Ks  = (bf16*)(smem + 16384);        // [128][64]
  bf16* VsT = (bf16*)(smem + 32768);        // [64][128]  (V transposed)
  bf16* Ps  = (bf16*)(smem + 49152);        // [128][128]

  int bIdx = blockIdx.x / NHc;
  int head = blockIdx.x % NHc;
  int tid = threadIdx.x, lane = tid & 31, w = tid >> 5;
  int hl = lane >> 4, nl = lane & 15;
  const size_t base = ((size_t)bIdx * Sc) * Hc + head * DHc;
  unsigned qsBase = lds_addr32(Qs), ksBase = lds_addr32(Ks);

  // Q and K: 128 rows x 64 bf16 = 128B per row = 8 x 16B chunks, async DMA
#pragma unroll
  for (int i = tid; i < 1024; i += 256) {
    int s_ = i >> 3, c = i & 7;
    unsigned lofs = (unsigned)(s_ * 128 + c * 16);
    const bf16* g = Qb + base + (size_t)s_ * Hc + c * 8;
    const bf16* g2 = Kb + base + (size_t)s_ * Hc + c * 8;
    async_copy_b128(qsBase + lofs, g);
    async_copy_b128(ksBase + lofs, g2);
  }
  for (int i = tid; i < 128 * 64; i += 256) {  // V needs transpose -> sync path
    int s_ = i >> 6, d = i & 63;
    VsT[(size_t)d * 128 + s_] = Vb[base + (size_t)s_ * Hc + d];
  }
  wait_async0();
  __syncthreads();

  // ---- S = Q K^T : each wave owns 16 query rows, scores in registers ----
  v8f sc[8];
#pragma unroll
  for (int nt = 0; nt < 8; ++nt) sc[nt] = (v8f){0.f,0.f,0.f,0.f,0.f,0.f,0.f,0.f};
#pragma unroll
  for (int kk = 0; kk < DHc; kk += 32) {
    v16bf a = load_frag_a(Qs, w * 16, 64, kk, lane);
#pragma unroll
    for (int nt = 0; nt < 8; ++nt) {
      v16bf bb = load_frag_b(Ks, nt * 16, 64, kk, lane);  // K row-major == K^T transposed
      sc[nt] = wmma_bf16(a, bb, sc[nt]);
    }
  }

  // ---- softmax (scale 1/8, mask bias); row stats via shfl within 16-lane group ----
  float rowmax[8], rowsum[8];
#pragma unroll
  for (int r = 0; r < 8; ++r) rowmax[r] = -3.4e38f;
#pragma unroll
  for (int nt = 0; nt < 8; ++nt) {
    int key = nt * 16 + nl;
    float biasv = (1.0f - maskp[(size_t)bIdx * Sc + key]) * -1e9f;
#pragma unroll
    for (int r = 0; r < 8; ++r) {
      float v = sc[nt][r] * 0.125f + biasv;
      sc[nt][r] = v;
      rowmax[r] = fmaxf(rowmax[r], v);
    }
  }
#pragma unroll
  for (int r = 0; r < 8; ++r)
    for (int off = 1; off < 16; off <<= 1)
      rowmax[r] = fmaxf(rowmax[r], __shfl_xor(rowmax[r], off, 32));
#pragma unroll
  for (int r = 0; r < 8; ++r) rowsum[r] = 0.f;
#pragma unroll
  for (int nt = 0; nt < 8; ++nt) {
    int key = nt * 16 + nl;
#pragma unroll
    for (int r = 0; r < 8; ++r) {
      float p = __expf(sc[nt][r] - rowmax[r]);
      rowsum[r] += p;
      Ps[(size_t)(w * 16 + hl * 8 + r) * 128 + key] = (bf16)p;  // unnormalized
    }
  }
#pragma unroll
  for (int r = 0; r < 8; ++r)
    for (int off = 1; off < 16; off <<= 1)
      rowsum[r] += __shfl_xor(rowsum[r], off, 32);
  __syncthreads();

  // ---- ctx = P V, normalize by rowsum in epilogue ----
  v16bf pa[4];
#pragma unroll
  for (int kk = 0; kk < 4; ++kk) pa[kk] = load_frag_a(Ps, w * 16, 128, kk * 32, lane);
#pragma unroll
  for (int nt = 0; nt < 4; ++nt) {
    v8f a2 = (v8f){0.f,0.f,0.f,0.f,0.f,0.f,0.f,0.f};
#pragma unroll
    for (int kk = 0; kk < 4; ++kk) {
      v16bf bb = load_frag_b(VsT, nt * 16, 128, kk * 32, lane);
      a2 = wmma_bf16(pa[kk], bb, a2);
    }
    int d = head * DHc + nt * 16 + nl;
#pragma unroll
    for (int r = 0; r < 8; ++r) {
      int qrow = bIdx * Sc + w * 16 + hl * 8 + r;
      ctx[(size_t)qrow * Hc + d] = (bf16)(a2[r] / rowsum[r]);
    }
  }
}

// ---------------- GNN kernels ----------------
__global__ void k_edge_feat(const int* __restrict__ attr, const float* __restrict__ ew,
                            const float* __restrict__ irf, const float* __restrict__ emb,
                            const int* __restrict__ ei, float* __restrict__ ef,
                            float* __restrict__ segs, float* __restrict__ cnt) {
  int j = blockIdx.x * blockDim.x + threadIdx.x;
  if (j >= Ec) return;
  int a = attr[j];
  float w = irf[a] * ew[j];
  int dst = ei[Ec + j];
#pragma unroll
  for (int d = 0; d < EDc; ++d) {
    float v = emb[(size_t)a * EDc + d] * w;
    ef[(size_t)j * EDc + d] = v;
    atomicAdd(segs + (size_t)dst * EDc + d, v);
  }
  atomicAdd(cnt + dst, 1.0f);
}

__global__ void k_mean(float* __restrict__ segs, const float* __restrict__ cnt) {
  int i = blockIdx.x * blockDim.x + threadIdx.x;
  if (i >= Nc) return;
  float c = fmaxf(cnt[i], 1.0f);
#pragma unroll
  for (int d = 0; d < EDc; ++d) segs[(size_t)i * EDc + d] /= c;
}

__global__ void k_init_gout(float* __restrict__ gout, const float* __restrict__ gb) {
  size_t i = (size_t)blockIdx.x * blockDim.x + threadIdx.x;
  if (i < (size_t)Nc * Hc) gout[i] = gb[i % Hc];
}

// one wave per edge: logit = sum_h lrelu(xl[src]+xr[dst]+ef@We)*att ; atomic max
__global__ void k_gat_logit(const int* __restrict__ ei, const float* __restrict__ xl,
                            const float* __restrict__ xr, const float* __restrict__ ef,
                            const float* __restrict__ meanef, const float* __restrict__ We,
                            const float* __restrict__ att, float* __restrict__ logit,
                            unsigned* __restrict__ mxkey) {
  int j = (blockIdx.x * blockDim.x + threadIdx.x) >> 5;
  int lane = threadIdx.x & 31;
  if (j >= E2c) return;
  int src, dst;
  const float* efj;
  if (j < Ec) { src = ei[j]; dst = ei[Ec + j]; efj = ef + (size_t)j * EDc; }
  else        { src = dst = j - Ec;            efj = meanef + (size_t)(j - Ec) * EDc; }
  float e16[EDc];
#pragma unroll
  for (int d = 0; d < EDc; ++d) e16[d] = efj[d];
  float acc = 0.f;
  for (int h = lane; h < Hc; h += 32) {
    float v = xl[(size_t)src * Hc + h] + xr[(size_t)dst * Hc + h];
#pragma unroll
    for (int d = 0; d < EDc; ++d) v += e16[d] * We[(size_t)d * Hc + h];
    v = v > 0.f ? v : 0.2f * v;   // leaky_relu(0.2)
    acc += v * att[h];
  }
  for (int off = 1; off < 32; off <<= 1) acc += __shfl_xor(acc, off, 32);
  if (lane == 0) {
    logit[j] = acc;
    atomicMax(mxkey + dst, fenc(acc));
  }
}

__global__ void k_gat_pexp(const int* __restrict__ ei, const float* __restrict__ logit,
                           const unsigned* __restrict__ mxkey, float* __restrict__ pexp,
                           float* __restrict__ den) {
  int j = blockIdx.x * blockDim.x + threadIdx.x;
  if (j >= E2c) return;
  int dst = (j < Ec) ? ei[Ec + j] : (j - Ec);
  float p = __expf(logit[j] - fdec(mxkey[dst]));
  pexp[j] = p;
  atomicAdd(den + dst, p);
}

__global__ void k_gat_scatter(const int* __restrict__ ei, const float* __restrict__ pexp,
                              const float* __restrict__ den, const float* __restrict__ xl,
                              float* __restrict__ gout) {
  int j = (blockIdx.x * blockDim.x + threadIdx.x) >> 5;
  int lane = threadIdx.x & 31;
  if (j >= E2c) return;
  int src, dst;
  if (j < Ec) { src = ei[j]; dst = ei[Ec + j]; }
  else        { src = dst = j - Ec; }
  float alpha = pexp[j] / den[dst];
  for (int h = lane; h < Hc; h += 32)
    atomicAdd(gout + (size_t)dst * Hc + h, alpha * xl[(size_t)src * Hc + h]);
}

__global__ void k_add_relu(const float* __restrict__ a, const float* __restrict__ b,
                           float* __restrict__ o, size_t n) {
  size_t i = (size_t)blockIdx.x * blockDim.x + threadIdx.x;
  if (i < n) o[i] = fmaxf(a[i] + b[i], 0.f);
}

__global__ void k_gather_cls(const float* __restrict__ x2, bf16* __restrict__ clsb) {
  int i = blockIdx.x * blockDim.x + threadIdx.x;
  if (i >= Bc * Hc) return;
  int b = i / Hc, h = i % Hc;
  clsb[i] = (bf16)x2[(size_t)(b * Sc) * Hc + h];
}

__global__ void k_cls_out(const float* __restrict__ hmid, const float* __restrict__ W2,
                          const float* __restrict__ b2, float* __restrict__ out) {
  int o = (blockIdx.x * blockDim.x + threadIdx.x) >> 5;
  int lane = threadIdx.x & 31;
  if (o >= Bc * 2) return;
  int bi = o >> 1, c = o & 1;
  float acc = 0.f;
  for (int k = lane; k < Hc / 2; k += 32)
    acc += hmid[(size_t)bi * (Hc / 2) + k] * W2[(size_t)k * 2 + c];
  for (int off = 1; off < 32; off <<= 1) acc += __shfl_xor(acc, off, 32);
  if (!lane) out[o] = acc + b2[c];
}

// ---------------- host launcher ----------------
extern "C" void kernel_launch(void* const* d_in, const int* in_sizes, int n_in,
                              void* d_out, int out_size, void* d_ws, size_t ws_size,
                              hipStream_t stream) {
  (void)in_sizes; (void)n_in; (void)out_size; (void)ws_size;
  const float* p_word   = (const float*)d_in[0];
  const float* p_pos    = (const float*)d_in[1];
  const float* p_tok    = (const float*)d_in[2];
  const float* p_elng   = (const float*)d_in[3];
  const float* p_elnb   = (const float*)d_in[4];
  const float* p_Wq     = (const float*)d_in[5];
  const float* p_bq     = (const float*)d_in[6];
  const float* p_Wk     = (const float*)d_in[7];
  const float* p_bk     = (const float*)d_in[8];
  const float* p_Wv     = (const float*)d_in[9];
  const float* p_bv     = (const float*)d_in[10];
  const float* p_Wo     = (const float*)d_in[11];
  const float* p_bo     = (const float*)d_in[12];
  const float* p_ln1g   = (const float*)d_in[13];
  const float* p_ln1b   = (const float*)d_in[14];
  const float* p_W1     = (const float*)d_in[15];
  const float* p_b1     = (const float*)d_in[16];
  const float* p_W2     = (const float*)d_in[17];
  const float* p_b2     = (const float*)d_in[18];
  const float* p_ln2g   = (const float*)d_in[19];
  const float* p_ln2b   = (const float*)d_in[20];
  const float* p_gWl    = (const float*)d_in[21];
  const float* p_gbl    = (const float*)d_in[22];
  const float* p_gWr    = (const float*)d_in[23];
  const float* p_gbr    = (const float*)d_in[24];
  const float* p_gWe    = (const float*)d_in[25];
  const float* p_gatt   = (const float*)d_in[26];
  const float* p_gbias  = (const float*)d_in[27];
  const float* p_eemb   = (const float*)d_in[28];
  const float* p_cW1    = (const float*)d_in[29];
  const float* p_cb1    = (const float*)d_in[30];
  const float* p_cW2    = (const float*)d_in[31];
  const float* p_cb2    = (const float*)d_in[32];
  const int*   in_ids   = (const int*)d_in[33];
  const float* in_mask  = (const float*)d_in[34];
  const int*   in_ei    = (const int*)d_in[35];
  const int*   in_eattr = (const int*)d_in[36];
  const float* in_ew    = (const float*)d_in[37];
  const float* in_irf   = (const float*)d_in[38];
  float* out = (float*)d_out;

  // workspace carve-up
  char* ws = (char*)d_ws;
  size_t off = 0;
  auto alloc = [&](size_t bytes) -> char* {
    char* p = ws + off;
    off += (bytes + 255) & ~(size_t)255;
    return p;
  };
  const size_t szHH = (size_t)Lc * Hc * Hc, szHF = (size_t)Lc * Hc * FFc;
  bf16* wq  = (bf16*)alloc(szHH * 2);
  bf16* wk  = (bf16*)alloc(szHH * 2);
  bf16* wv  = (bf16*)alloc(szHH * 2);
  bf16* wo  = (bf16*)alloc(szHH * 2);
  bf16* w1  = (bf16*)alloc(szHF * 2);
  bf16* w2  = (bf16*)alloc(szHF * 2);
  bf16* gwl = (bf16*)alloc((size_t)Hc * Hc * 2);
  bf16* gwr = (bf16*)alloc((size_t)Hc * Hc * 2);
  bf16* cw1 = (bf16*)alloc((size_t)Hc * (Hc / 2) * 2);
  float* h    = (float*)alloc((size_t)Nc * Hc * 4);
  float* tmp  = (float*)alloc((size_t)Nc * Hc * 4);
  bf16*  hbf  = (bf16*)alloc((size_t)Nc * Hc * 2);
  bf16*  qb   = (bf16*)alloc((size_t)Nc * Hc * 2);
  bf16*  kb   = (bf16*)alloc((size_t)Nc * Hc * 2);
  bf16*  vb   = (bf16*)alloc((size_t)Nc * Hc * 2);
  bf16*  ctxb = (bf16*)alloc((size_t)Nc * Hc * 2);
  bf16*  ffb  = (bf16*)alloc((size_t)Nc * FFc * 2);
  float* xl   = (float*)alloc((size_t)Nc * Hc * 4);
  float* xr   = (float*)alloc((size_t)Nc * Hc * 4);
  float* gout = (float*)alloc((size_t)Nc * Hc * 4);
  float* ef   = (float*)alloc((size_t)Ec * EDc * 4);
  float* segs = (float*)alloc((size_t)Nc * EDc * 4);
  float* cnt  = (float*)alloc((size_t)Nc * 4);
  float* logit = (float*)alloc((size_t)E2c * 4);
  float* pexp  = (float*)alloc((size_t)E2c * 4);
  unsigned* mxkey = (unsigned*)alloc((size_t)Nc * 4);
  float* den  = (float*)alloc((size_t)Nc * 4);
  bf16*  clsb = (bf16*)alloc((size_t)Bc * Hc * 2);
  float* hmid = (float*)alloc((size_t)Bc * (Hc / 2) * 4);

  auto conv = [&](const float* s, bf16* d, size_t n) {
    k_convert_bf16<<<(unsigned)((n + 255) / 256), 256, 0, stream>>>(s, d, n);
  };

  // 1) weights -> bf16 (one streaming pass; negligible vs WMMA time)
  conv(p_Wq, wq, szHH); conv(p_Wk, wk, szHH); conv(p_Wv, wv, szHH); conv(p_Wo, wo, szHH);
  conv(p_W1, w1, szHF); conv(p_W2, w2, szHF);
  conv(p_gWl, gwl, (size_t)Hc * Hc); conv(p_gWr, gwr, (size_t)Hc * Hc);
  conv(p_cW1, cw1, (size_t)Hc * (Hc / 2));

  // 2) embeddings + LN
  k_embed<<<(Nc * Hc + 255) / 256, 256, 0, stream>>>(in_ids, p_word, p_pos, p_tok, tmp);
  k_layernorm<<<Nc, 256, 0, stream>>>(tmp, p_elng, p_elnb, h, hbf, Hc);

  // 3) encoder layers
  for (int l = 0; l < Lc; ++l) {
    const size_t wofs = (size_t)l * Hc * Hc, fofs = (size_t)l * Hc * FFc;
    // QKV: bias, no resid, no act, bf16 out
    launch_gemm<0,false,false,true>(hbf, wq + wofs, p_bq + l * Hc, nullptr,
                                    nullptr, qb, Nc, Hc, Hc, stream);
    launch_gemm<0,false,false,true>(hbf, wk + wofs, p_bk + l * Hc, nullptr,
                                    nullptr, kb, Nc, Hc, Hc, stream);
    launch_gemm<0,false,false,true>(hbf, wv + wofs, p_bv + l * Hc, nullptr,
                                    nullptr, vb, Nc, Hc, Hc, stream);
    attn_kernel<<<Bc * NHc, 256, 81920, stream>>>(qb, kb, vb, in_mask, ctxb);
    // Wo: bias + residual(h), f32 out
    launch_gemm<0,true,true,false>(ctxb, wo + wofs, p_bo + l * Hc, h,
                                   tmp, nullptr, Nc, Hc, Hc, stream);
    k_layernorm<<<Nc, 256, 0, stream>>>(tmp, p_ln1g + l * Hc, p_ln1b + l * Hc, h, hbf, Hc);
    // FF1: bias + GELU, bf16 out
    launch_gemm<1,false,false,true>(hbf, w1 + fofs, p_b1 + l * FFc, nullptr,
                                    nullptr, ffb, Nc, FFc, Hc, stream);
    // FF2: bias + residual(h), f32 out
    launch_gemm<0,true,true,false>(ffb, w2 + fofs, p_b2 + l * Hc, h,
                                   tmp, nullptr, Nc, Hc, FFc, stream);
    k_layernorm<<<Nc, 256, 0, stream>>>(tmp, p_ln2g + l * Hc, p_ln2b + l * Hc, h, hbf, Hc);
  }

  // 4) GNN
  k_zero_f32<<<((size_t)Nc * EDc + 255) / 256, 256, 0, stream>>>(segs, (size_t)Nc * EDc);
  k_zero_f32<<<(Nc + 255) / 256, 256, 0, stream>>>(cnt, Nc);
  k_zero_f32<<<(Nc + 255) / 256, 256, 0, stream>>>(den, Nc);
  k_zero_u32<<<(Nc + 255) / 256, 256, 0, stream>>>(mxkey, Nc);
  k_edge_feat<<<(Ec + 255) / 256, 256, 0, stream>>>(in_eattr, in_ew, in_irf, p_eemb,
                                                    in_ei, ef, segs, cnt);
  k_mean<<<(Nc + 255) / 256, 256, 0, stream>>>(segs, cnt);
  launch_gemm<0,false,true,false>(hbf, gwl, p_gbl, nullptr, xl, nullptr,
                                  Nc, Hc, Hc, stream);
  launch_gemm<0,false,true,false>(hbf, gwr, p_gbr, nullptr, xr, nullptr,
                                  Nc, Hc, Hc, stream);
  k_init_gout<<<((size_t)Nc * Hc + 255) / 256, 256, 0, stream>>>(gout, p_gbias);
  {
    unsigned blocks = (E2c * 32 + 255) / 256;
    k_gat_logit<<<blocks, 256, 0, stream>>>(in_ei, xl, xr, ef, segs, p_gWe, p_gatt,
                                            logit, mxkey);
    k_gat_pexp<<<(E2c + 255) / 256, 256, 0, stream>>>(in_ei, logit, mxkey, pexp, den);
    k_gat_scatter<<<blocks, 256, 0, stream>>>(in_ei, pexp, den, xl, gout);
  }
  k_add_relu<<<((size_t)Nc * Hc + 255) / 256, 256, 0, stream>>>(gout, h, tmp,
                                                                (size_t)Nc * Hc);

  // 5) classifier head
  k_gather_cls<<<(Bc * Hc + 255) / 256, 256, 0, stream>>>(tmp, clsb);
  launch_gemm<2,false,true,false>(clsb, cw1, p_cb1, nullptr, hmid, nullptr,
                                  Bc, Hc / 2, Hc, stream);
  k_cls_out<<<8, 256, 0, stream>>>(hmid, p_cW2, p_cb2, out);
}
